// RNN_7627861917718
// MI455X (gfx1250) — matrix operand order repaired and checked by reading, not compiled, not run on hardware
//
#include <hip/hip_runtime.h>
#include <hip/hip_bf16.h>

typedef __bf16 bf16;
typedef __attribute__((ext_vector_type(16))) __bf16 v16bf;
typedef __attribute__((ext_vector_type(8)))  float  v8f;
typedef __attribute__((ext_vector_type(4)))  float  vf4;

#define BB 256
#define TT 512
#define II 128
#define HH 128
#define CC 10

__device__ __forceinline__ float fast_tanh(float x) {
    // tanh(x) = 1 - 2/(exp(2x)+1); exp via exp2: 2*log2(e) = 2.885390081777927
    float t = __builtin_amdgcn_exp2f(x * 2.885390081777927f);
    return 1.0f - 2.0f * __builtin_amdgcn_rcpf(t + 1.0f);
}

// ---------------- fp32 -> bf16 convert ----------------
__global__ void cvt_f32_bf16(const float* __restrict__ in, bf16* __restrict__ out, int n) {
    int i = (blockIdx.x * blockDim.x + threadIdx.x) * 4;
    if (i >= n) return;
    vf4 v = *(const vf4*)(in + i);
    union { bf16 b[4]; unsigned long long u; } o;
    o.b[0] = (bf16)v.x; o.b[1] = (bf16)v.y; o.b[2] = (bf16)v.z; o.b[3] = (bf16)v.w;
    *(unsigned long long*)(out + i) = o.u;
}

// ---------------- bias sums: b_ih + b_hh per layer ----------------
__global__ void bias_sum(const float* a0, const float* b0,
                         const float* a1, const float* b1, float* out) {
    int i = threadIdx.x;            // 256 threads
    if (i < 128) out[i] = a0[i] + b0[i];
    else         out[i] = a1[i - 128] + b1[i - 128];
}

// ------ input-projection GEMM: P[t*B + b, :] = A[b*T + t, :] * W^T + bias -------
// A bf16 row-major [M=B*T, K=128]; W bf16 row-major [N=128, K=128];
// Out fp32, t-major [T, B, 128] so the scan reads contiguous per-step tiles.
__global__ __launch_bounds__(128) void gemm_proj(
    const bf16* __restrict__ A, const bf16* __restrict__ W,
    const float* __restrict__ bias, float* __restrict__ Out) {
    __shared__ bf16 wlds[128 * 128];
    {   // stage W into LDS: 32 KB = 2048 x 16B
        const vf4* src = (const vf4*)W;
        vf4* dst = (vf4*)wlds;
        for (int idx = threadIdx.x; idx < 2048; idx += 128) dst[idx] = src[idx];
    }
    __syncthreads();

    const int lane = threadIdx.x & 31;
    const int wave = threadIdx.x >> 5;
    const int m  = lane & 15;        // row-in-tile (A) / col (B,D)
    const int hi = lane >> 4;        // lane half
    const long row0 = (long)blockIdx.x * 64 + wave * 16;

    // A strip: 16 rows x 128 K, native 16-bit A layout (K-contiguous 16B chunks)
    v16bf a[4];
    const bf16* arow = A + (row0 + m) * 128;
#pragma unroll
    for (int kc = 0; kc < 4; ++kc) {
        union { v16bf v; vf4 f[2]; } u;
        u.f[0] = *(const vf4*)(arow + kc * 32 + hi * 8);
        u.f[1] = *(const vf4*)(arow + kc * 32 + 16 + hi * 8);
        a[kc] = u.v;
    }

    v8f acc[8];
#pragma unroll
    for (int nt = 0; nt < 8; ++nt) acc[nt] = v8f{};

#pragma unroll
    for (int kc = 0; kc < 4; ++kc) {
        v16bf bmat[8];                      // batched B-tile loads -> one wait/chunk
#pragma unroll
        for (int nt = 0; nt < 8; ++nt) {
            const bf16* wrow = wlds + (nt * 16 + m) * 128 + kc * 32 + hi * 8;
            union { v16bf v; vf4 f[2]; } u;
            u.f[0] = *(const vf4*)(wrow);
            u.f[1] = *(const vf4*)(wrow + 16);
            bmat[nt] = u.v;
        }
        __builtin_amdgcn_sched_barrier(0);  // keep all 16 ds_loads ahead of WMMAs
#pragma unroll
        for (int nt = 0; nt < 8; ++nt)
            acc[nt] = __builtin_amdgcn_wmma_f32_16x16x32_bf16(
                false, a[kc], false, bmat[nt], (short)0, acc[nt], false, false);
        __builtin_amdgcn_sched_barrier(0);
    }

#pragma unroll
    for (int nt = 0; nt < 8; ++nt) {
        const int col = nt * 16 + m;
        const float bv = bias[col];
#pragma unroll
        for (int j = 0; j < 8; ++j) {
            const long ar = row0 + hi * 8 + j;               // = b*512 + t
            const long orow = (ar & 511) * BB + (ar >> 9);   // t*B + b
            Out[orow * 128 + col] = acc[nt][j] + bv;
        }
    }
}

// ---------------- recurrence: h_t = tanh(P_t + h_{t-1} W_hh^T) ----------------
// 4 waves per workgroup share one 16-batch-row tile; each wave owns 32 output
// cols and keeps its W_hh slice (64 VGPRs) in registers for the whole scan:
// the 512-step serial chain contains NO weight loads at all.
__global__ __launch_bounds__(128) void rnn_scan(
    const float* __restrict__ P,      // [T, B, 128] pre-activations (t-major)
    const bf16* __restrict__ Whh,     // [128,128] bf16 row-major [N,K]
    bf16* __restrict__ h_seq,         // [B*T,128] bf16 (row = b*T+t) or nullptr
    float* __restrict__ h_last) {     // [B,128] fp32 or nullptr
    __shared__ bf16 hlds[16 * 128];

    const int tid  = threadIdx.x;
    const int lane = tid & 31;
    const int wave = tid >> 5;        // 0..3 -> owns nt = 2*wave, 2*wave+1
    const int m  = lane & 15;
    const int hi = lane >> 4;
    const int b0 = blockIdx.x * 16;

    // W_hh slice for this wave's two N-tiles, resident in registers
    v16bf wreg[2][4];
#pragma unroll
    for (int q = 0; q < 2; ++q) {
        const int nt = wave * 2 + q;
#pragma unroll
        for (int kc = 0; kc < 4; ++kc) {
            const bf16* wrow = Whh + (nt * 16 + m) * 128 + kc * 32 + hi * 8;
            union { v16bf v; vf4 f[2]; } u;
            u.f[0] = *(const vf4*)(wrow);
            u.f[1] = *(const vf4*)(wrow + 16);
            wreg[q][kc] = u.v;
        }
    }
    {   // h0 = 0: 4 KB over 128 threads
        vf4 z = {0.f, 0.f, 0.f, 0.f};
        vf4* hz = (vf4*)hlds;
        for (int idx = tid; idx < 256; idx += 128) hz[idx] = z;
    }
    __syncthreads();

    for (int t = 0; t < TT; ++t) {
        const float* Pt = P + (size_t)t * BB * 128 + (size_t)b0 * 128;
        if (t + 1 < TT)   // pull next step's contiguous 8KB pre-act tile into cache
            __builtin_prefetch(Pt + (size_t)BB * 128 + tid * 16, 0, 0);

        // A = h_{t-1} tile from LDS, native 16-bit A layout (each wave full copy)
        v16bf a[4];
#pragma unroll
        for (int kc = 0; kc < 4; ++kc) {
            const bf16* hrow = hlds + m * 128 + kc * 32 + hi * 8;
            union { v16bf v; vf4 f[2]; } u;
            u.f[0] = *(const vf4*)(hrow);
            u.f[1] = *(const vf4*)(hrow + 16);
            a[kc] = u.v;
        }

        // 8 WMMAs, weights already in registers: 2 chains of depth 4
        v8f acc[2];
        acc[0] = v8f{}; acc[1] = v8f{};
#pragma unroll
        for (int kc = 0; kc < 4; ++kc) {
#pragma unroll
            for (int q = 0; q < 2; ++q)
                acc[q] = __builtin_amdgcn_wmma_f32_16x16x32_bf16(
                    false, a[kc], false, wreg[q][kc], (short)0, acc[q], false, false);
        }

        __syncthreads();   // WAR: everyone has read h_{t-1} before it is overwritten

        // h_t = tanh(pre + acc); write this wave's 32 cols back to LDS
#pragma unroll
        for (int q = 0; q < 2; ++q) {
            const int col = (wave * 2 + q) * 16 + m;
#pragma unroll
            for (int j = 0; j < 8; ++j) {
                const int r = j + hi * 8;
                const float v = fast_tanh(acc[q][j] + Pt[r * 128 + col]);
                hlds[r * 128 + col] = (bf16)v;
                if (h_last && t == TT - 1) h_last[(b0 + r) * 128 + col] = v;
            }
        }
        __syncthreads();   // RAW: h_t complete before next-step reads / dump

        if (h_seq) {       // coalesced dump of 16x128 bf16 tile (2 x b128 per thread)
            for (int idx = tid; idx < 256; idx += 128) {
                const int r = idx >> 4, c = idx & 15;
                *(vf4*)(h_seq + ((long)(b0 + r) * TT + t) * 128 + c * 8) =
                    ((const vf4*)hlds)[idx];
            }
        }
    }
}

// ---------------- tiny classifier: out[B,10] = h2 @ Wfc^T + bfc ----------------
__global__ void fc_kernel(const float* __restrict__ h, const float* __restrict__ Wfc,
                          const float* __restrict__ bfc, float* __restrict__ out) {
    int idx = blockIdx.x * blockDim.x + threadIdx.x;
    if (idx >= BB * CC) return;
    int b = idx / CC, c = idx % CC;
    const vf4* hv = (const vf4*)(h + b * 128);
    const vf4* wv = (const vf4*)(Wfc + c * 128);
    float s = bfc[c];
#pragma unroll
    for (int q = 0; q < 32; ++q) {
        vf4 a = hv[q], w = wv[q];
        s += a.x * w.x + a.y * w.y + a.z * w.z + a.w * w.w;
    }
    out[idx] = s;
}

extern "C" void kernel_launch(void* const* d_in, const int* in_sizes, int n_in,
                              void* d_out, int out_size, void* d_ws, size_t ws_size,
                              hipStream_t stream) {
    const float* x     = (const float*)d_in[0];
    const float* W_ih0 = (const float*)d_in[1];
    const float* W_hh0 = (const float*)d_in[2];
    const float* b_ih0 = (const float*)d_in[3];
    const float* b_hh0 = (const float*)d_in[4];
    const float* W_ih1 = (const float*)d_in[5];
    const float* W_hh1 = (const float*)d_in[6];
    const float* b_ih1 = (const float*)d_in[7];
    const float* b_hh1 = (const float*)d_in[8];
    const float* W_fc  = (const float*)d_in[9];
    const float* b_fc  = (const float*)d_in[10];
    float* out = (float*)d_out;

    char* ws = (char*)d_ws;
    const size_t BT = (size_t)BB * TT;
    bf16*  xb   = (bf16*)(ws);                              // BT*128 bf16 = 32 MB
    bf16*  h1b  = (bf16*)(ws + BT * 128 * 2);               // BT*128 bf16 = 32 MB
    float* P    = (float*)(ws + BT * 128 * 4);              // BT*128 f32  = 64 MB
    bf16*  wb   = (bf16*)(ws + BT * 128 * 8);               // 4 x 128*128 bf16
    float* bsum = (float*)(ws + BT * 128 * 8 + 4 * 16384 * 2);
    float* h2   = (float*)((char*)bsum + 256 * 4);          // B*128 f32
    bf16* wb_ih0 = wb;
    bf16* wb_hh0 = wb + 16384;
    bf16* wb_ih1 = wb + 2 * 16384;
    bf16* wb_hh1 = wb + 3 * 16384;

    // 1) precision conversion (x + the four 128x128 weights)
    {
        int n = (int)(BT * II);
        cvt_f32_bf16<<<(n / 4 + 255) / 256, 256, 0, stream>>>(x, xb, n);
        cvt_f32_bf16<<<16, 256, 0, stream>>>(W_ih0, wb_ih0, 16384);
        cvt_f32_bf16<<<16, 256, 0, stream>>>(W_hh0, wb_hh0, 16384);
        cvt_f32_bf16<<<16, 256, 0, stream>>>(W_ih1, wb_ih1, 16384);
        cvt_f32_bf16<<<16, 256, 0, stream>>>(W_hh1, wb_hh1, 16384);
        bias_sum<<<1, 256, 0, stream>>>(b_ih0, b_hh0, b_ih1, b_hh1, bsum);
    }
    // 2) layer 0 input projection (parallel GEMM over all B*T rows)
    gemm_proj<<<(int)(BT / 64), 128, 0, stream>>>(xb, wb_ih0, bsum, P);
    // 3) layer 0 recurrence -> h1 sequence (bf16)
    rnn_scan<<<BB / 16, 128, 0, stream>>>(P, wb_hh0, h1b, nullptr);
    // 4) layer 1 input projection (reuses P)
    gemm_proj<<<(int)(BT / 64), 128, 0, stream>>>(h1b, wb_ih1, bsum + 128, P);
    // 5) layer 1 recurrence -> final hidden state (fp32)
    rnn_scan<<<BB / 16, 128, 0, stream>>>(P, wb_hh1, nullptr, h2);
    // 6) classifier
    fc_kernel<<<(BB * CC + 127) / 128, 128, 0, stream>>>(h2, W_fc, b_fc, out);
}
